// Model_84164179133293
// MI455X (gfx1250) — matrix operand order repaired and checked by reading, not compiled, not run on hardware
//
#include <hip/hip_runtime.h>
#include <hip/hip_bf16.h>

typedef __attribute__((ext_vector_type(16))) _Float16 v16h;
typedef __attribute__((ext_vector_type(8)))  float    v8f;
typedef __attribute__((ext_vector_type(4)))  unsigned int v4u;
typedef __attribute__((ext_vector_type(8)))  int      v8i;
typedef __attribute__((ext_vector_type(4)))  int      v4i;

#if defined(__has_builtin)
#if __has_builtin(__builtin_amdgcn_tensor_load_to_lds) && __has_builtin(__builtin_amdgcn_s_wait_tensorcnt)
#define HAVE_TDM 1
#endif
#endif
#ifndef HAVE_TDM
#define HAVE_TDM 0
#endif

#define B_    32
#define L_    512
#define C_    21
#define D_    512
#define H_    8
#define DK_   64
#define DFF_  2048
#define NL_   3
#define COUT_ 21
#define BL_   (B_ * L_)

// ---------------------------------------------------------------- stats ----
__global__ void stats_kernel(const float* __restrict__ x, float* __restrict__ means,
                             float* __restrict__ stdev) {
  int b = blockIdx.x;
  int t = threadIdx.x;  // block of 32
  __shared__ float varr[32];
  varr[t] = 0.f;
  if (t < C_) {
    const float* xb = x + (size_t)b * L_ * C_ + t;
    float s = 0.f;
    for (int l = 0; l < L_; ++l) s += xb[(size_t)l * C_];
    float m = s / (float)L_;
    means[b * C_ + t] = m;
    float v = 0.f;
    for (int l = 0; l < L_; ++l) { float d = xb[(size_t)l * C_] - m; v += d * d; }
    varr[t] = v;
  }
  __syncthreads();
  if (t == 0) {
    float v = 0.f;
    for (int c = 0; c < C_; ++c) v += varr[c];
    stdev[b] = sqrtf(v / (float)(L_ * C_) + 1e-5f);
  }
}

// ---------------------------------------------------------- tau / delta ----
__global__ void tau_delta_kernel(const float* __restrict__ means, const float* __restrict__ stdev,
    const float* tw1, const float* tb1, const float* tw2, const float* tb2,
    const float* tw3, const float* tb3,
    const float* dw1, const float* db1, const float* dw2, const float* db2,
    const float* dw3, const float* db3,
    float* __restrict__ tau, float* __restrict__ delta) {
  int b = blockIdx.x, t = threadIdx.x;  // block 256
  __shared__ float in1[48];
  __shared__ float h1[D_];
  __shared__ float h2[D_];
  __shared__ float red[256];
  if (t < C_) in1[1 + t] = means[b * C_ + t];
  if (t == 0) in1[0] = stdev[b];
  __syncthreads();
  // --- tau path ---
  for (int d = t; d < D_; d += 256) {
    float s = tb1[d];
    for (int i = 0; i < C_ + 1; ++i) s += in1[i] * tw1[i * D_ + d];
    h1[d] = fmaxf(s, 0.f);
  }
  __syncthreads();
  for (int d = t; d < D_; d += 256) {
    float s = tb2[d];
    for (int i = 0; i < D_; ++i) s += h1[i] * tw2[i * D_ + d];
    h2[d] = fmaxf(s, 0.f);
  }
  __syncthreads();
  {
    float s = 0.f;
    for (int d = t; d < D_; d += 256) s += h2[d] * tw3[d];
    red[t] = s;
    __syncthreads();
    for (int k = 128; k > 0; k >>= 1) { if (t < k) red[t] += red[t + k]; __syncthreads(); }
    if (t == 0) tau[b] = __expf(red[0] + tb3[0]);
  }
  __syncthreads();
  // --- delta path (delta_in = [means, x_mean] and x_mean == means over L) ---
  if (t < C_) { float m = means[b * C_ + t]; in1[t] = m; in1[C_ + t] = m; }
  __syncthreads();
  for (int d = t; d < D_; d += 256) {
    float s = db1[d];
    for (int i = 0; i < 2 * C_; ++i) s += in1[i] * dw1[i * D_ + d];
    h1[d] = fmaxf(s, 0.f);
  }
  __syncthreads();
  for (int d = t; d < D_; d += 256) {
    float s = db2[d];
    for (int i = 0; i < D_; ++i) s += h1[i] * dw2[i * D_ + d];
    h2[d] = fmaxf(s, 0.f);
  }
  __syncthreads();
  for (int l = t; l < L_; l += 256) {
    float s = db3[l];
    for (int d = 0; d < D_; ++d) s += h2[d] * dw3[d * L_ + l];
    delta[b * L_ + l] = s;
  }
}

// ---------------------------------------------------------- embedding -----
__global__ void embed_kernel(const float* __restrict__ xenc, const float* __restrict__ xmark,
                             const float* __restrict__ convw, const float* __restrict__ tw,
                             const float* __restrict__ means, const float* __restrict__ stdev,
                             float* __restrict__ xf, _Float16* __restrict__ xh) {
  int bl = blockIdx.x;
  int b = bl >> 9;
  int l = bl & (L_ - 1);
  int t = threadIdx.x;  // block 256
  __shared__ float xs[3 * C_];
  __shared__ float mk[4];
  float rstd = 1.f / stdev[b];
  if (t < 3 * C_) {
    int kk = t / C_, c = t - kk * C_;
    int ll = (l + kk - 1 + L_) & (L_ - 1);  // circular pad
    xs[t] = (xenc[((size_t)b * L_ + ll) * C_ + c] - means[b * C_ + c]) * rstd;
  }
  if (t < 4) mk[t] = xmark[((size_t)b * L_ + l) * 4 + t];
  __syncthreads();
  const float LOG1E4_D = 9.210340371976184f / (float)D_;
  for (int d = t; d < D_; d += 256) {
    const float* w = convw + (size_t)d * (C_ * 3);
    float acc = 0.f;
    for (int c = 0; c < C_; ++c)
      for (int kk = 0; kk < 3; ++kk)
        acc += w[c * 3 + kk] * xs[kk * C_ + c];
    for (int j = 0; j < 4; ++j) acc += mk[j] * tw[j * D_ + d];
    int i2 = (d >> 1) << 1;
    float freq = __expf(-(float)i2 * LOG1E4_D);
    float ang = (float)l * freq;
    acc += (d & 1) ? __cosf(ang) : __sinf(ang);
    xf[(size_t)bl * D_ + d] = acc;
    xh[(size_t)bl * D_ + d] = (_Float16)acc;
  }
}

// --------------------------------------- f32[K,N] -> f16 transposed [N,K] --
__global__ void cvt_f16_t_kernel(const float* __restrict__ a, _Float16* __restrict__ o,
                                 int K, int N) {
  int i = blockIdx.x * 256 + threadIdx.x;
  if (i < K * N) {
    int k = i / N, n = i - k * N;
    o[(size_t)n * K + k] = (_Float16)a[i];
  }
}

// ---------------------------------------------------------- WMMA GEMM -----
__device__ __forceinline__ float gelu_f(float x) {
  float x3 = x * x * x;
  return 0.5f * x * (1.f + tanhf(0.7978845608028654f * (x + 0.044715f * x3)));
}

__device__ __forceinline__ void store_tile(v8f acc, int rowBase, int col,
    const float* __restrict__ bias, float* __restrict__ outF, _Float16* __restrict__ outH,
    long long cOff, int ldc, int act) {
  float bv = bias ? bias[col] : 0.f;
#pragma unroll
  for (int i = 0; i < 8; ++i) {
    float val = acc[i] + bv;
    if (act == 1) val = gelu_f(val);
    long long off = cOff + (long long)(rowBase + i) * ldc + col;
    if (outF) outF[off] = val;
    if (outH) outH[off] = (_Float16)val;
  }
}

// A fragment: lane 0-15 rows, contiguous K order (k = 16*(lane/16) + s), matching B.
__device__ __forceinline__ v16h load_a_frag(const _Float16* __restrict__ base, int lane) {
  int hiA = (lane >> 4) << 4;
  v16h a;
#pragma unroll
  for (int v = 0; v < 8; ++v) {
    int kb = hiA + (v << 1);
    a[2 * v] = base[kb];
    a[2 * v + 1] = base[kb + 1];
  }
  return a;
}
// B fragment: Bt[n][k] staged; K striped across lane halves.
__device__ __forceinline__ v16h load_b_frag(const _Float16* __restrict__ base) {
  v16h b;
#pragma unroll
  for (int e = 0; e < 16; ++e) b[e] = base[e];
  return b;
}

#if HAVE_TDM
// TDM: 2-D tile (rowlen=32 halfs) x rows, global row stride = stride elements,
// into LDS with 8-DWORD pad per 16-DWORD row => 48-half LDS row stride.
__device__ __forceinline__ void tdm_load_2d(unsigned int lds_addr, const void* gaddr,
                                            int rows, int stride_el,
                                            int tensor_d0, int tensor_d1) {
  unsigned long long ga = (unsigned long long)(uintptr_t)gaddr;
  v4u g0;
  g0[0] = 1u;                                   // count=1 (valid), user mode
  g0[1] = lds_addr;                             // lds_addr (bytes)
  g0[2] = (unsigned int)(ga & 0xFFFFFFFFu);     // global_addr[31:0]
  g0[3] = (unsigned int)((ga >> 32) & 0x01FFFFFFu) | (2u << 30);  // addr[56:32] | type=2
  v8i g1;
  // data_size=2B(code1), pad_enable, pad_interval=16dw(code3), pad_amount=8dw(code7)
  g1[0] = (1 << 16) | (1 << 20) | (3 << 22) | (7 << 25);
  g1[1] = (int)(((unsigned)tensor_d0 & 0xFFFFu) << 16);                    // td0[15:0]
  g1[2] = (int)((((unsigned)tensor_d0 >> 16) & 0xFFFFu) |
                (((unsigned)tensor_d1 & 0xFFFFu) << 16));                  // td0 hi | td1 lo
  g1[3] = (int)((((unsigned)tensor_d1 >> 16) & 0xFFFFu) | (32u << 16));    // td1 hi | tile_dim0=32
  g1[4] = rows;                                 // tile_dim1 (tile_dim2=0)
  g1[5] = stride_el;                            // tensor_dim0_stride[31:0]
  g1[6] = 0;
  g1[7] = 0;
  v4i z4 = {0, 0, 0, 0};
  v8i z8 = {0, 0, 0, 0, 0, 0, 0, 0};
  __builtin_amdgcn_tensor_load_to_lds(g0, g1, z4, z4, z8, 0);
}
#endif

// ---- NT GEMM: C[M,N] = A[M,K] * B^T (B given [N,K] row-major). Tile 64x128,
// 8 waves x (32x32) = 4 WMMA/wave/K-step. TDM double-buffered staging. ----
__global__ __launch_bounds__(256) void gemm_nt_kernel(
    const _Float16* __restrict__ A, long long aOuter, long long aInner, int lda,
    const _Float16* __restrict__ Bm, long long bOuter, long long bInner, int ldb,
    const float* __restrict__ bias,
    float* __restrict__ outF, _Float16* __restrict__ outH,
    long long cOuter, long long cInner, int ldc,
    int M, int N, int K, int sub, int act) {
  __shared__ __align__(16) _Float16 As[2][64 * 48];
  __shared__ __align__(16) _Float16 Bs[2][128 * 48];
  int z = blockIdx.z;
  int zo = z / sub, zi = z - zo * sub;
  int blockM = blockIdx.y * 64;
  int blockN = blockIdx.x * 128;
  const _Float16* Ag = A + (long long)zo * aOuter + (long long)zi * aInner +
                       (long long)blockM * lda;
  const _Float16* Bg = Bm + (long long)zo * bOuter + (long long)zi * bInner +
                       (long long)blockN * ldb;
  long long cOff = (long long)zo * cOuter + (long long)zi * cInner;
  int t = threadIdx.x;
  int lane = t & 31;
  int wv = t >> 5;
  int wm = wv & 1;   // 2 row-slices of 32
  int wn = wv >> 1;  // 4 col-slices of 32
  v8f acc00 = {}, acc01 = {}, acc10 = {}, acc11 = {};
  int KT = K >> 5;
#if HAVE_TDM
  bool issuer = (wv == 0);
  if (issuer) {
    tdm_load_2d((unsigned int)(uintptr_t)&As[0][0], Ag, 64, lda, lda, 64);
    tdm_load_2d((unsigned int)(uintptr_t)&Bs[0][0], Bg, 128, ldb, ldb, 128);
  }
#endif
  for (int kt = 0; kt < KT; ++kt) {
    int buf = kt & 1;
    __syncthreads();  // all reads of the buffer we are about to overwrite are done
#if HAVE_TDM
    bool more = (kt + 1 < KT);
    if (issuer) {
      if (more) {
        tdm_load_2d((unsigned int)(uintptr_t)&As[buf ^ 1][0], Ag + ((kt + 1) << 5),
                    64, lda, lda, 64);
        tdm_load_2d((unsigned int)(uintptr_t)&Bs[buf ^ 1][0], Bg + ((kt + 1) << 5),
                    128, ldb, ldb, 128);
        __builtin_amdgcn_s_wait_tensorcnt(2);
      } else {
        __builtin_amdgcn_s_wait_tensorcnt(0);
      }
    }
#else
    {  // manual staging fallback
      int k0 = kt << 5;
      int aRow = t >> 2, aKc = (t & 3) << 3;
      *reinterpret_cast<uint4*>(&As[buf][aRow * 48 + aKc]) =
          *reinterpret_cast<const uint4*>(Ag + (size_t)aRow * lda + k0 + aKc);
      int n = t >> 1, kc = (t & 1) << 4;
      *reinterpret_cast<uint4*>(&Bs[buf][n * 48 + kc]) =
          *reinterpret_cast<const uint4*>(Bg + (size_t)n * ldb + k0 + kc);
      *reinterpret_cast<uint4*>(&Bs[buf][n * 48 + kc + 8]) =
          *reinterpret_cast<const uint4*>(Bg + (size_t)n * ldb + k0 + kc + 8);
    }
#endif
    __syncthreads();  // tile kt visible in LDS to all waves
    int row = lane & 15;
    const _Float16* Ab = As[buf];
    const _Float16* Bb = Bs[buf];
    v16h a0 = load_a_frag(&Ab[(wm * 32 + row) * 48], lane);
    v16h a1 = load_a_frag(&Ab[(wm * 32 + 16 + row) * 48], lane);
    int kB = (lane >> 4) << 4;
    v16h b0 = load_b_frag(&Bb[(wn * 32 + row) * 48 + kB]);
    v16h b1 = load_b_frag(&Bb[(wn * 32 + 16 + row) * 48 + kB]);
    acc00 = __builtin_amdgcn_wmma_f32_16x16x32_f16(false, a0, false, b0, (short)0, acc00, false, false);
    acc01 = __builtin_amdgcn_wmma_f32_16x16x32_f16(false, a0, false, b1, (short)0, acc01, false, false);
    acc10 = __builtin_amdgcn_wmma_f32_16x16x32_f16(false, a1, false, b0, (short)0, acc10, false, false);
    acc11 = __builtin_amdgcn_wmma_f32_16x16x32_f16(false, a1, false, b1, (short)0, acc11, false, false);
  }
  int col0 = blockN + wn * 32 + (lane & 15);
  int row0 = blockM + wm * 32 + ((lane >> 4) << 3);
  store_tile(acc00, row0, col0, bias, outF, outH, cOff, ldc, act);
  store_tile(acc01, row0, col0 + 16, bias, outF, outH, cOff, ldc, act);
  store_tile(acc10, row0 + 16, col0, bias, outF, outH, cOff, ldc, act);
  store_tile(acc11, row0 + 16, col0 + 16, bias, outF, outH, cOff, ldc, act);
}

// ---- NN GEMM (B given [K,N]); 64x64 tile, used for attn x V (N=64). ----
__global__ __launch_bounds__(256) void gemm_nn_kernel(
    const _Float16* __restrict__ A, long long aOuter, long long aInner, int lda,
    const _Float16* __restrict__ Bm, long long bOuter, long long bInner, int ldb,
    const float* __restrict__ bias,
    float* __restrict__ outF, _Float16* __restrict__ outH,
    long long cOuter, long long cInner, int ldc,
    int M, int N, int K, int sub, int act) {
  __shared__ __align__(16) _Float16 As[64 * 48];
  __shared__ __align__(16) _Float16 Bs[64 * 48];
  int z = blockIdx.z;
  int zo = z / sub, zi = z - zo * sub;
  const _Float16* Ag = A + (long long)zo * aOuter + (long long)zi * aInner;
  const _Float16* Bg = Bm + (long long)zo * bOuter + (long long)zi * bInner;
  long long cOff = (long long)zo * cOuter + (long long)zi * cInner;
  int blockM = blockIdx.y * 64;
  int blockN = blockIdx.x * 64;
  int t = threadIdx.x;
  int lane = t & 31;
  int wv = t >> 5;
  int wm = wv & 3;
  int wn = wv >> 2;
  v8f acc0 = {};
  v8f acc1 = {};
  int aRow = t >> 2, aKc = (t & 3) << 3;
  for (int k0 = 0; k0 < K; k0 += 32) {
    __syncthreads();
    *reinterpret_cast<uint4*>(&As[aRow * 48 + aKc]) = *reinterpret_cast<const uint4*>(
        Ag + (size_t)(blockM + aRow) * lda + k0 + aKc);
    if (k0 + 32 < K)
      __builtin_prefetch(Ag + (size_t)(blockM + aRow) * lda + k0 + 32 + aKc, 0, 0);
    {  // B[K,N] -> Bt[n][k] transpose staging
      int kk = t & 31, ng = t >> 5;
      uint4 v = *reinterpret_cast<const uint4*>(
          Bg + (size_t)(k0 + kk) * ldb + blockN + (ng << 3));
      _Float16 tmp[8];
      *reinterpret_cast<uint4*>(tmp) = v;
#pragma unroll
      for (int j = 0; j < 8; ++j) Bs[((ng << 3) + j) * 48 + kk] = tmp[j];
    }
    __syncthreads();
    int row = lane & 15;
    v16h a = load_a_frag(&As[(wm * 16 + row) * 48], lane);
    int kB = (lane >> 4) << 4;
    v16h b0 = load_b_frag(&Bs[(wn * 32 + row) * 48 + kB]);
    v16h b1 = load_b_frag(&Bs[(wn * 32 + 16 + row) * 48 + kB]);
    acc0 = __builtin_amdgcn_wmma_f32_16x16x32_f16(false, a, false, b0, (short)0, acc0, false, false);
    acc1 = __builtin_amdgcn_wmma_f32_16x16x32_f16(false, a, false, b1, (short)0, acc1, false, false);
  }
  int colBase = blockN + wn * 32 + (lane & 15);
  int rowBase = blockM + wm * 16 + ((lane >> 4) << 3);
  store_tile(acc0, rowBase, colBase, bias, outF, outH, cOff, ldc, act);
  store_tile(acc1, rowBase, colBase + 16, bias, outF, outH, cOff, ldc, act);
}

// ----------------------------------------------------------- softmax ------
__global__ void softmax_kernel(const float* __restrict__ scores, _Float16* __restrict__ attn,
                               const float* __restrict__ tau, const float* __restrict__ delta) {
  int l = blockIdx.x, h = blockIdx.y, b = blockIdx.z;
  int t = threadIdx.x;  // block 256, row of 512
  size_t rowOff = ((((size_t)b * H_ + h) * L_) + l) * L_;
  const float* srow = scores + rowOff;
  _Float16* arow = attn + rowOff;
  float ts = tau[b] * 0.125f;  // tau * 1/sqrt(64)
  const float* del = delta + b * L_;
  __shared__ float red[256];
  float v0 = ts * srow[t] + del[t];
  float v1 = ts * srow[t + 256] + del[t + 256];
  red[t] = fmaxf(v0, v1);
  __syncthreads();
  for (int k = 128; k > 0; k >>= 1) { if (t < k) red[t] = fmaxf(red[t], red[t + k]); __syncthreads(); }
  float mx = red[0];
  __syncthreads();
  float e0 = __expf(v0 - mx), e1 = __expf(v1 - mx);
  red[t] = e0 + e1;
  __syncthreads();
  for (int k = 128; k > 0; k >>= 1) { if (t < k) red[t] += red[t + k]; __syncthreads(); }
  float rs = 1.f / red[0];
  arow[t] = (_Float16)(e0 * rs);
  arow[t + 256] = (_Float16)(e1 * rs);
}

// ------------------------------------------- residual + layernorm ---------
__global__ void resid_ln_kernel(const float* __restrict__ x, const float* __restrict__ y,
                                const float* __restrict__ g, const float* __restrict__ bta,
                                float* __restrict__ xo, _Float16* __restrict__ xho) {
  size_t row = blockIdx.x;
  int t = threadIdx.x;  // block 256, D=512
  const float* xr = x + row * D_;
  float r0 = xr[t], r1 = xr[t + 256];
  if (y) { const float* yr = y + row * D_; r0 += yr[t]; r1 += yr[t + 256]; }
  __shared__ float red[256];
  red[t] = r0 + r1;
  __syncthreads();
  for (int k = 128; k > 0; k >>= 1) { if (t < k) red[t] += red[t + k]; __syncthreads(); }
  float mean = red[0] / (float)D_;
  __syncthreads();
  float d0 = r0 - mean, d1 = r1 - mean;
  red[t] = d0 * d0 + d1 * d1;
  __syncthreads();
  for (int k = 128; k > 0; k >>= 1) { if (t < k) red[t] += red[t + k]; __syncthreads(); }
  float rinv = rsqrtf(red[0] / (float)D_ + 1e-5f);
  float o0 = g[t] * d0 * rinv + bta[t];
  float o1 = g[t + 256] * d1 * rinv + bta[t + 256];
  xo[row * D_ + t] = o0;
  xo[row * D_ + t + 256] = o1;
  xho[row * D_ + t] = (_Float16)o0;
  xho[row * D_ + t + 256] = (_Float16)o1;
}

// -------------------------------------- final projection + de-norm --------
__global__ void proj_kernel(const float* __restrict__ x, const float* __restrict__ pw,
                            const float* __restrict__ pb, const float* __restrict__ stdev,
                            const float* __restrict__ means, float* __restrict__ out) {
  size_t row = blockIdx.x;
  int co = threadIdx.x;  // block 32
  if (co >= COUT_) return;
  int b = (int)(row >> 9);
  const float* xr = x + row * D_;
  float acc = pb[co];
  for (int d = 0; d < D_; ++d) acc += xr[d] * pw[d * COUT_ + co];
  out[row * COUT_ + co] = acc * stdev[b] + means[b * C_ + co];
}

// ===========================================================================
extern "C" void kernel_launch(void* const* d_in, const int* in_sizes, int n_in,
                              void* d_out, int out_size, void* d_ws, size_t ws_size,
                              hipStream_t stream) {
  (void)in_sizes; (void)n_in; (void)out_size; (void)ws_size;
  const float* x_enc  = (const float*)d_in[0];
  const float* x_mark = (const float*)d_in[1];
  const float* convw  = (const float*)d_in[2];
  const float* tempw  = (const float*)d_in[3];
  const float* tw1 = (const float*)d_in[4];  const float* tb1 = (const float*)d_in[5];
  const float* tw2 = (const float*)d_in[6];  const float* tb2 = (const float*)d_in[7];
  const float* tw3 = (const float*)d_in[8];  const float* tb3 = (const float*)d_in[9];
  const float* dw1 = (const float*)d_in[10]; const float* db1 = (const float*)d_in[11];
  const float* dw2 = (const float*)d_in[12]; const float* db2 = (const float*)d_in[13];
  const float* dw3 = (const float*)d_in[14]; const float* db3 = (const float*)d_in[15];
  const float* proj_w = (const float*)d_in[16];
  const float* proj_b = (const float*)d_in[17];
  const float* ln_g = (const float*)d_in[18];
  const float* ln_b = (const float*)d_in[19];

  // ---- workspace carve (256B aligned) ----
  char* base = (char*)d_ws;
  size_t off = 0;
  auto alloc = [&](size_t bytes) -> void* {
    off = (off + 255) & ~(size_t)255;
    void* p = base + off;
    off += bytes;
    return p;
  };
  float*    means  = (float*)alloc(B_ * C_ * 4);
  float*    stdev  = (float*)alloc(B_ * 4);
  float*    tau    = (float*)alloc(B_ * 4);
  float*    delta  = (float*)alloc((size_t)B_ * L_ * 4);
  float*    xf     = (float*)alloc((size_t)BL_ * D_ * 4);
  float*    yf     = (float*)alloc((size_t)BL_ * D_ * 4);
  _Float16* xh     = (_Float16*)alloc((size_t)BL_ * D_ * 2);
  _Float16* qh     = (_Float16*)alloc((size_t)BL_ * D_ * 2);
  _Float16* kh     = (_Float16*)alloc((size_t)BL_ * D_ * 2);
  _Float16* vh     = (_Float16*)alloc((size_t)BL_ * D_ * 2);
  _Float16* oh     = (_Float16*)alloc((size_t)BL_ * D_ * 2);
  _Float16* hh     = (_Float16*)alloc((size_t)BL_ * DFF_ * 2);
  float*    scores = (float*)alloc((size_t)B_ * H_ * L_ * L_ * 4);
  _Float16* attn   = (_Float16*)alloc((size_t)B_ * H_ * L_ * L_ * 2);
  _Float16 *wqh[NL_], *wkh[NL_], *wvh[NL_], *woh[NL_], *c1wh[NL_], *c2wh[NL_];
  for (int li = 0; li < NL_; ++li) {
    wqh[li]  = (_Float16*)alloc((size_t)D_ * D_ * 2);
    wkh[li]  = (_Float16*)alloc((size_t)D_ * D_ * 2);
    wvh[li]  = (_Float16*)alloc((size_t)D_ * D_ * 2);
    woh[li]  = (_Float16*)alloc((size_t)D_ * D_ * 2);
    c1wh[li] = (_Float16*)alloc((size_t)D_ * DFF_ * 2);
    c2wh[li] = (_Float16*)alloc((size_t)DFF_ * D_ * 2);
  }

  // ---- weight conversion f32[K,N] -> f16 transposed [N,K] ----
  auto cvt_t = [&](const float* src, _Float16* dst, int K, int N) {
    cvt_f16_t_kernel<<<(K * N + 255) / 256, 256, 0, stream>>>(src, dst, K, N);
  };
  for (int li = 0; li < NL_; ++li) {
    const int pb0 = 20 + li * 16;
    cvt_t((const float*)d_in[pb0 + 0], wqh[li], D_, D_);
    cvt_t((const float*)d_in[pb0 + 2], wkh[li], D_, D_);
    cvt_t((const float*)d_in[pb0 + 4], wvh[li], D_, D_);
    cvt_t((const float*)d_in[pb0 + 6], woh[li], D_, D_);
    cvt_t((const float*)d_in[pb0 + 8], c1wh[li], D_, DFF_);   // -> [DFF, D]
    cvt_t((const float*)d_in[pb0 + 10], c2wh[li], DFF_, D_);  // -> [D, DFF]
  }

  // ---- stationarization stats, tau/delta, embedding ----
  stats_kernel<<<B_, 32, 0, stream>>>(x_enc, means, stdev);
  tau_delta_kernel<<<B_, 256, 0, stream>>>(means, stdev, tw1, tb1, tw2, tb2, tw3, tb3,
                                           dw1, db1, dw2, db2, dw3, db3, tau, delta);
  embed_kernel<<<BL_, 256, 0, stream>>>(x_enc, x_mark, convw, tempw, means, stdev, xf, xh);

  auto gemm_nt = [&](const _Float16* A, long long aO, long long aI, int lda,
                     const _Float16* Bm, long long bO, long long bI, int ldb,
                     const float* bias, float* oF, _Float16* oH,
                     long long cO, long long cI, int ldc,
                     int M, int N, int K, int batch, int sub, int act) {
    dim3 g(N / 128, M / 64, batch);
    gemm_nt_kernel<<<g, 256, 0, stream>>>(A, aO, aI, lda, Bm, bO, bI, ldb,
                                          bias, oF, oH, cO, cI, ldc, M, N, K, sub, act);
  };
  auto gemm_nn = [&](const _Float16* A, long long aO, long long aI, int lda,
                     const _Float16* Bm, long long bO, long long bI, int ldb,
                     const float* bias, float* oF, _Float16* oH,
                     long long cO, long long cI, int ldc,
                     int M, int N, int K, int batch, int sub, int act) {
    dim3 g(N / 64, M / 64, batch);
    gemm_nn_kernel<<<g, 256, 0, stream>>>(A, aO, aI, lda, Bm, bO, bI, ldb,
                                          bias, oF, oH, cO, cI, ldc, M, N, K, sub, act);
  };

  const long long LD = (long long)L_ * D_;   // per-batch stride in [B,L,D]
  const long long LL = (long long)L_ * L_;   // per-(b,h) score tile
  const long long HLL = (long long)H_ * LL;

  // ---- encoder layers ----
  for (int li = 0; li < NL_; ++li) {
    const int pb0 = 20 + li * 16;
    const float* bq = (const float*)d_in[pb0 + 1];
    const float* bk = (const float*)d_in[pb0 + 3];
    const float* bv = (const float*)d_in[pb0 + 5];
    const float* bo = (const float*)d_in[pb0 + 7];
    const float* c1b = (const float*)d_in[pb0 + 9];
    const float* c2b = (const float*)d_in[pb0 + 11];
    const float* ln1g = (const float*)d_in[pb0 + 12];
    const float* ln1b = (const float*)d_in[pb0 + 13];
    const float* ln2g = (const float*)d_in[pb0 + 14];
    const float* ln2b = (const float*)d_in[pb0 + 15];

    // Q/K/V projections: [BL,512] x Wt[512,512]^T
    gemm_nt(xh, 0, 0, D_, wqh[li], 0, 0, D_, bq, nullptr, qh, 0, 0, D_, BL_, D_, D_, 1, 1, 0);
    gemm_nt(xh, 0, 0, D_, wkh[li], 0, 0, D_, bk, nullptr, kh, 0, 0, D_, BL_, D_, D_, 1, 1, 0);
    gemm_nt(xh, 0, 0, D_, wvh[li], 0, 0, D_, bv, nullptr, vh, 0, 0, D_, BL_, D_, D_, 1, 1, 0);

    // scores[b,h] = Q[b,:,h,:] * K[b,:,h,:]^T  (batched over b*h, K=64)
    gemm_nt(qh, LD, DK_, D_,
            kh, LD, DK_, D_,
            nullptr, scores, nullptr, HLL, LL, L_,
            L_, L_, DK_, B_ * H_, H_, 0);

    softmax_kernel<<<dim3(L_, H_, B_), 256, 0, stream>>>(scores, attn, tau, delta);

    // O[b,:,h,:] = attn[b,h] * V[b,:,h,:]  (batched, N=64, B given [K,N])
    gemm_nn(attn, HLL, LL, L_,
            vh, LD, DK_, D_,
            nullptr, nullptr, oh, LD, DK_, D_,
            L_, DK_, L_, B_ * H_, H_, 0);

    // output projection + residual + LN1
    gemm_nt(oh, 0, 0, D_, woh[li], 0, 0, D_, bo, yf, nullptr, 0, 0, D_, BL_, D_, D_, 1, 1, 0);
    resid_ln_kernel<<<BL_, 256, 0, stream>>>(xf, yf, ln1g, ln1b, xf, xh);

    // FFN
    gemm_nt(xh, 0, 0, D_, c1wh[li], 0, 0, D_, c1b, nullptr, hh, 0, 0, DFF_,
            BL_, DFF_, D_, 1, 1, 1 /*gelu*/);
    gemm_nt(hh, 0, 0, DFF_, c2wh[li], 0, 0, DFF_, c2b, yf, nullptr, 0, 0, D_,
            BL_, D_, DFF_, 1, 1, 0);
    resid_ln_kernel<<<BL_, 256, 0, stream>>>(xf, yf, ln2g, ln2b, xf, xh);
  }

  // ---- final LN + projection + de-normalization ----
  resid_ln_kernel<<<BL_, 256, 0, stream>>>(xf, nullptr, ln_g, ln_b, xf, xh);
  proj_kernel<<<BL_, 32, 0, stream>>>(xf, proj_w, proj_b, stdev, means, (float*)d_out);
}
